// grica_Encoder_68908455297219
// MI455X (gfx1250) — compile-verified, matrix operand order checked by reading, not run on hardware
//
#include <hip/hip_runtime.h>

#define N_DIN   256
#define N_DOUT  64
#define BM      128      // rows of x per block (cov and out)
#define TSTRIDE 136      // transposed-y LDS row stride in bf16 (272B: conflict-free)
#define NSLOTS  64       // slotted atomic accumulators for C partials

typedef __bf16 v16bf __attribute__((ext_vector_type(16)));
typedef __bf16 v8bf  __attribute__((ext_vector_type(8)));
typedef float  v8f   __attribute__((ext_vector_type(8)));
typedef float  v4f   __attribute__((ext_vector_type(4)));

// Native hardware conversions (v_cvt_pk_bf16_f32 / v_fma_mix_f32_bf16 on gfx1250).
static __device__ __forceinline__ __bf16 f2bf(float f) { return (__bf16)f; }
static __device__ __forceinline__ float bf2f(__bf16 b) { return (float)b; }

// Per-lane 16-element bf16 fragment (split hi+lo) from a row-major f32 row of
// length 256. Element e of lane (lm, hi16): k = kb*32 + (e/8)*16 + hi16*8 + (e%8)
// NT=true -> non-temporal loads (streaming data that must not pollute L2).
template <bool NT>
static __device__ __forceinline__ void load_frag_f32row(
    const float* __restrict__ rowptr, int kb, int hi16, bool ok,
    v16bf& fh, v16bf& fl) {
  float v[16];
  if (ok) {
    const float* p = rowptr + kb * 32 + hi16 * 8;
    v4f a0, a1, b0, b1;
    if constexpr (NT) {
      a0 = __builtin_nontemporal_load((const v4f*)(p));
      a1 = __builtin_nontemporal_load((const v4f*)(p + 4));
      b0 = __builtin_nontemporal_load((const v4f*)(p + 16));
      b1 = __builtin_nontemporal_load((const v4f*)(p + 20));
    } else {
      a0 = *(const v4f*)(p);
      a1 = *(const v4f*)(p + 4);
      b0 = *(const v4f*)(p + 16);
      b1 = *(const v4f*)(p + 20);
    }
#pragma unroll
    for (int e = 0; e < 4; ++e) {
      v[e]      = a0[e];
      v[e + 4]  = a1[e];
      v[e + 8]  = b0[e];
      v[e + 12] = b1[e];
    }
  } else {
#pragma unroll
    for (int e = 0; e < 16; ++e) v[e] = 0.f;
  }
#pragma unroll
  for (int e = 0; e < 16; ++e) {
    __bf16 h = f2bf(v[e]);
    fh[e] = h;
    fl[e] = f2bf(v[e] - bf2f(h));
  }
}

// Fragment from transposed-y LDS: elements 0..7 / 8..15 are two contiguous
// 8x-bf16 (16B) runs -> two ds_load_b128.
static __device__ __forceinline__ v16bf load_frag_ylds(
    const __bf16* __restrict__ buf, int col, int kb, int hi16) {
  const int base = col * TSTRIDE + kb * 32 + hi16 * 8;
  v8bf p0 = *(const v8bf*)&buf[base];
  v8bf p1 = *(const v8bf*)&buf[base + 16];
  return __builtin_shufflevector(p0, p1, 0, 1, 2, 3, 4, 5, 6, 7,
                                 8, 9, 10, 11, 12, 13, 14, 15);
}

// split-precision product: (ah+al)(bh+bl) ~= ah*bh + ah*bl + al*bh
static __device__ __forceinline__ v8f wmma3(v8f acc, v16bf ah, v16bf al,
                                            v16bf bh, v16bf bl) {
  acc = __builtin_amdgcn_wmma_f32_16x16x32_bf16(false, ah, false, bh, (short)0, acc, false, false);
  acc = __builtin_amdgcn_wmma_f32_16x16x32_bf16(false, ah, false, bl, (short)0, acc, false, false);
  acc = __builtin_amdgcn_wmma_f32_16x16x32_bf16(false, al, false, bh, (short)0, acc, false, false);
  return acc;
}

__global__ void k_zero(float* __restrict__ p, int n) {
  int i = blockIdx.x * blockDim.x + threadIdx.x;
  if (i < n) p[i] = 0.f;
}

// Fused: y_block = x_block @ Wfc^T (WMMA, split bf16) -> transposed LDS, then
// C_partial = y_block^T y_block (WMMA) -> slotted atomic accumulation.
__global__ __launch_bounds__(128) void k_cov(const float* __restrict__ x,
                                             const float* __restrict__ Wfc,
                                             float* __restrict__ slots,
                                             int nrows) {
  __shared__ __bf16 yth[N_DOUT * TSTRIDE];  // y^T, hi terms: [col][row]
  __shared__ __bf16 ytl[N_DOUT * TSTRIDE];  // y^T, lo terms
  const int tid  = threadIdx.x;
  const int wave = tid >> 5;
  const int lane = tid & 31;
  const int lm   = lane & 15;
  const int hi16 = lane >> 4;
  const int row0 = blockIdx.x * BM;

  // phase 1: y tiles; wave owns row tiles rt = wave and wave+4
  v8f acc[2][4] = {};
  for (int kb = 0; kb < N_DIN / 32; ++kb) {
    v16bf bh[4], bl[4];
#pragma unroll
    for (int nt = 0; nt < 4; ++nt)  // B[k,n] = Wfc[n,k]  (cached: regular loads)
      load_frag_f32row<false>(Wfc + (nt * 16 + lm) * N_DIN, kb, hi16, true, bh[nt], bl[nt]);
#pragma unroll
    for (int ri = 0; ri < 2; ++ri) {
      const int grow = row0 + (wave + ri * 4) * 16 + lm;
      const bool ok  = grow < nrows;
      v16bf ah, al;  // x is streamed once: non-temporal
      load_frag_f32row<true>(x + (size_t)grow * N_DIN, kb, hi16, ok, ah, al);
#pragma unroll
      for (int nt = 0; nt < 4; ++nt)
        acc[ri][nt] = wmma3(acc[ri][nt], ah, al, bh[nt], bl[nt]);
    }
  }
  // store y^T: lane's 8 rows (r + hi16*8) are contiguous -> packed b128 stores
#pragma unroll
  for (int ri = 0; ri < 2; ++ri)
#pragma unroll
    for (int nt = 0; nt < 4; ++nt) {
      v8bf h8, l8;
#pragma unroll
      for (int r = 0; r < 8; ++r) {
        float yv = acc[ri][nt][r];
        __bf16 h = f2bf(yv);
        h8[r] = h;
        l8[r] = f2bf(yv - bf2f(h));
      }
      const int col  = nt * 16 + lm;
      const int base = col * TSTRIDE + (wave + ri * 4) * 16 + hi16 * 8;
      *(v8bf*)&yth[base] = h8;
      *(v8bf*)&ytl[base] = l8;
    }
  __syncthreads();

  // phase 2: C_partial = y^T y ; wave owns o1-tile == wave
  v8f cacc[4] = {};
  for (int kb = 0; kb < BM / 32; ++kb) {
    v16bf ah = load_frag_ylds(yth, wave * 16 + lm, kb, hi16);
    v16bf al = load_frag_ylds(ytl, wave * 16 + lm, kb, hi16);
#pragma unroll
    for (int o2 = 0; o2 < 4; ++o2) {
      v16bf bh = load_frag_ylds(yth, o2 * 16 + lm, kb, hi16);
      v16bf bl = load_frag_ylds(ytl, o2 * 16 + lm, kb, hi16);
      cacc[o2] = wmma3(cacc[o2], ah, al, bh, bl);
    }
  }
  float* slot = slots + (blockIdx.x & (NSLOTS - 1)) * (N_DOUT * N_DOUT);
#pragma unroll
  for (int o2 = 0; o2 < 4; ++o2)
#pragma unroll
    for (int r = 0; r < 8; ++r) {
      int orow = wave * 16 + r + hi16 * 8;
      int ocol = o2 * 16 + lm;
      atomicAdd(&slot[orow * N_DOUT + ocol], cacc[o2][r]);
    }
}

__global__ void k_reduce(const float* __restrict__ slots, float* __restrict__ C0,
                         float n) {
  int i = blockIdx.x * blockDim.x + threadIdx.x;
  if (i < N_DOUT * N_DOUT) {
    float s = 0.f;
    for (int sl = 0; sl < NSLOTS; ++sl) s += slots[sl * N_DOUT * N_DOUT + i];
    C0[i] = s / n;
  }
}

// Symmetric 64x64 matvec for lane t's column-pair (2t,2t+1):
// a = sum_k C[2t][k] v[k], b = sum_k C[2t+1][k] v[k], using C[k][2t:2t+2] loads.
// 4 independent FMA chains per output + float4 v broadcasts to cut latency.
static __device__ __forceinline__ float2 symm_matvec(
    const float* __restrict__ Cs, const float* __restrict__ vs, int t) {
  float a0 = 0.f, a1 = 0.f, a2 = 0.f, a3 = 0.f;
  float b0 = 0.f, b1 = 0.f, b2 = 0.f, b3 = 0.f;
#pragma unroll
  for (int k = 0; k < 64; k += 4) {
    v4f v4 = *(const v4f*)&vs[k];
    float2 c0 = *(const float2*)&Cs[(k + 0) * 64 + 2 * t];
    float2 c1 = *(const float2*)&Cs[(k + 1) * 64 + 2 * t];
    float2 c2 = *(const float2*)&Cs[(k + 2) * 64 + 2 * t];
    float2 c3 = *(const float2*)&Cs[(k + 3) * 64 + 2 * t];
    a0 = fmaf(c0.x, v4[0], a0);
    b0 = fmaf(c0.y, v4[0], b0);
    a1 = fmaf(c1.x, v4[1], a1);
    b1 = fmaf(c1.y, v4[1], b1);
    a2 = fmaf(c2.x, v4[2], a2);
    b2 = fmaf(c2.y, v4[2], b2);
    a3 = fmaf(c3.x, v4[3], a3);
    b3 = fmaf(c3.y, v4[3], b3);
  }
  float2 r;
  r.x = (a0 + a1) + (a2 + a3);
  r.y = (b0 + b1) + (b2 + b3);
  return r;
}

// Whole power-whitening scan in one wave32. C is symmetric, so lane t owns
// column-pair (2t,2t+1): row dot products read C[k][2t:2t+2] as 8B LDS loads.
__global__ __launch_bounds__(32) void k_power(const float* __restrict__ C0,
                                              const float* __restrict__ R,
                                              float* __restrict__ Wout) {
  __shared__ float Cs[64 * 64];
  __shared__ float Ws[64 * 64];
  __shared__ float vs[64];
  const int t = threadIdx.x;  // 0..31
  for (int i = t; i < 64 * 64; i += 32) { Cs[i] = C0[i]; Ws[i] = 0.f; }
  __syncthreads();
  for (int j = 0; j < 64; ++j) {
    vs[2 * t]     = R[(2 * t) * 64 + j];      // r_col = R[:, j]
    vs[2 * t + 1] = R[(2 * t + 1) * 64 + j];
    __syncthreads();
    for (int it = 0; it < 250; ++it) {
      float2 ab = symm_matvec(Cs, vs, t);
      float s = ab.x * ab.x + ab.y * ab.y;
#pragma unroll
      for (int off = 16; off > 0; off >>= 1) s += __shfl_xor(s, off, 32);
      float inv = rsqrtf(s);
      vs[2 * t]     = ab.x * inv;
      vs[2 * t + 1] = ab.y * inv;
      __syncthreads();
    }
    // dominant eigenvalue l = ||C v||
    float2 ab = symm_matvec(Cs, vs, t);
    float s = ab.x * ab.x + ab.y * ab.y;
#pragma unroll
    for (int off = 16; off > 0; off >>= 1) s += __shfl_xor(s, off, 32);
    float l = sqrtf(s);
    float rsl = rsqrtf(l);
    float v0 = vs[2 * t], v1 = vs[2 * t + 1];
#pragma unroll 4
    for (int k = 0; k < 64; ++k) {
      float vk = vs[k];
      float2 c2 = *(float2*)&Cs[k * 64 + 2 * t];
      c2.x = fmaf(-l * vk, v0, c2.x);
      c2.y = fmaf(-l * vk, v1, c2.y);
      *(float2*)&Cs[k * 64 + 2 * t] = c2;
      float2 w2 = *(float2*)&Ws[k * 64 + 2 * t];
      w2.x = fmaf(rsl * vk, v0, w2.x);
      w2.y = fmaf(rsl * vk, v1, w2.y);
      *(float2*)&Ws[k * 64 + 2 * t] = w2;
    }
    __syncthreads();
  }
  for (int i = t; i < 64 * 64; i += 32) Wout[i] = Ws[i];
}

// WT = W @ Wfc  (64 x 256): whitened = x @ WT^T, so y is never materialized.
__global__ void k_wt(const float* __restrict__ W, const float* __restrict__ Wfc,
                     float* __restrict__ WT) {
  int idx = blockIdx.x * blockDim.x + threadIdx.x;
  if (idx < N_DOUT * N_DIN) {
    int o = idx >> 8;
    int k = idx & 255;
    float s = 0.f;
#pragma unroll 8
    for (int jj = 0; jj < N_DOUT; ++jj)
      s = fmaf(W[o * N_DOUT + jj], Wfc[jj * N_DIN + k], s);
    WT[idx] = s;
  }
}

// whitened = x @ WT^T via split-bf16 WMMA; 128 rows per block, kb-outer so the
// WT (B) fragments are converted once and reused across both row tiles.
__global__ __launch_bounds__(128) void k_out(const float* __restrict__ x,
                                             const float* __restrict__ WT,
                                             float* __restrict__ out,
                                             int nrows) {
  const int tid  = threadIdx.x;
  const int wave = tid >> 5;
  const int lane = tid & 31;
  const int lm   = lane & 15;
  const int hi16 = lane >> 4;
  const int row0 = blockIdx.x * BM;

  v8f acc[2][4] = {};
  for (int kb = 0; kb < N_DIN / 32; ++kb) {
    v16bf bh[4], bl[4];
#pragma unroll
    for (int nt = 0; nt < 4; ++nt)  // WT cached in L2: regular loads
      load_frag_f32row<false>(WT + (nt * 16 + lm) * N_DIN, kb, hi16, true, bh[nt], bl[nt]);
#pragma unroll
    for (int ri = 0; ri < 2; ++ri) {
      const int grow = row0 + (wave + ri * 4) * 16 + lm;
      const bool ok  = grow < nrows;
      v16bf ah, al;  // x streamed: non-temporal
      load_frag_f32row<true>(x + (size_t)grow * N_DIN, kb, hi16, ok, ah, al);
#pragma unroll
      for (int nt = 0; nt < 4; ++nt)
        acc[ri][nt] = wmma3(acc[ri][nt], ah, al, bh[nt], bl[nt]);
    }
  }
#pragma unroll
  for (int ri = 0; ri < 2; ++ri)
#pragma unroll
    for (int nt = 0; nt < 4; ++nt)
#pragma unroll
      for (int r = 0; r < 8; ++r) {
        int orow = row0 + (wave + ri * 4) * 16 + r + hi16 * 8;
        if (orow < nrows)  // output streamed: non-temporal stores
          __builtin_nontemporal_store(acc[ri][nt][r],
                                      &out[(size_t)orow * N_DOUT + nt * 16 + lm]);
      }
}

extern "C" void kernel_launch(void* const* d_in, const int* in_sizes, int n_in,
                              void* d_out, int out_size, void* d_ws, size_t ws_size,
                              hipStream_t stream) {
  (void)n_in; (void)out_size; (void)ws_size;
  const float* x   = (const float*)d_in[0];
  const float* Wfc = (const float*)d_in[1];
  const float* R   = (const float*)d_in[2];
  float* out = (float*)d_out;
  const int nrows = in_sizes[0] / N_DIN;  // 500000

  float* ws    = (float*)d_ws;
  float* slots = ws;                                    // NSLOTS * 4096 floats
  float* C0    = slots + NSLOTS * N_DOUT * N_DOUT;      // 4096
  float* Wm    = C0 + N_DOUT * N_DOUT;                  // 4096
  float* WT    = Wm + N_DOUT * N_DOUT;                  // 16384

  const int zn = NSLOTS * N_DOUT * N_DOUT;
  k_zero<<<(zn + 255) / 256, 256, 0, stream>>>(slots, zn);
  const int nbcov = (nrows + BM - 1) / BM;
  k_cov<<<nbcov, 128, 0, stream>>>(x, Wfc, slots, nrows);
  k_reduce<<<(N_DOUT * N_DOUT + 255) / 256, 256, 0, stream>>>(slots, C0, (float)nrows);
  k_power<<<1, 32, 0, stream>>>(C0, R, Wm);
  k_wt<<<(N_DOUT * N_DIN + 255) / 256, 256, 0, stream>>>(Wm, Wfc, WT);
  k_out<<<(nrows + BM - 1) / BM, 128, 0, stream>>>(x, WT, out, nrows);
}
